// MultiHeadedAttention_52905407152492
// MI455X (gfx1250) — matrix operand order repaired and checked by reading, not compiled, not run on hardware
//
#include <hip/hip_runtime.h>

typedef _Float16 v16h __attribute__((ext_vector_type(16)));
typedef _Float16 v8h  __attribute__((ext_vector_type(8)));
typedef _Float16 v2h  __attribute__((ext_vector_type(2)));
typedef float    v8f  __attribute__((ext_vector_type(8)));

#define HEADSZ 64
#define NHEADS 12
#define DMODEL 768
#define BATCH  2
#define SEQ    2048
#define MROWS  (BATCH * SEQ)   /* 4096 */

// Build a v16h WMMA fragment from two contiguous 8-half (16B) runs.
__device__ __forceinline__ v16h ldfrag(const _Float16* p0, const _Float16* p1) {
  v8h lo = *(const v8h*)p0;
  v8h hi = *(const v8h*)p1;
  return __builtin_shufflevector(lo, hi, 0, 1, 2, 3, 4, 5, 6, 7,
                                          8, 9, 10, 11, 12, 13, 14, 15);
}

__device__ __forceinline__ v8f wmma_f16(v16h a, v16h b, v8f c) {
  // D = A(16x32 f16) * B(32x16 f16) + C(16x16 f32)
  return __builtin_amdgcn_wmma_f32_16x16x32_f16(false, a, false, b,
                                                (short)0, c, false, false);
}

// ---------------------------------------------------------------------------
// Tiled GEMM: C(M=4096, N=768) = A(4096x768) * W(768x768) + bias
// BM=128, BN=64, BK=32; 256 threads = 8 waves (4x2); each wave -> 32x32 tile
// (2x2 WMMA accumulators -> 4 WMMAs per K-step, register A/B reuse).
// A_HALF:   A operand is fp16 (else fp32, converted while staging to LDS)
// OUT_MODE: 0 = fp16 row-major, 1 = fp16 per-head transposed Vt[b][h][d][s],
//           2 = fp32 row-major (final output)
// ---------------------------------------------------------------------------
template <bool A_HALF, int OUT_MODE>
__global__ __launch_bounds__(256) void gemm16(const void* __restrict__ Av,
                                              const float* __restrict__ W,
                                              const float* __restrict__ bias,
                                              void* __restrict__ Out) {
  constexpr int BM = 128, BN = 64, BK = 32;
  constexpr int LDA = BK + 8;  // padded rows (80B) to spread LDS banks
  constexpr int LDB = BK + 8;
  __shared__ alignas(16) _Float16 sA[BM * LDA];   // A tile, row-major (m,k)
  __shared__ alignas(16) _Float16 sBt[BN * LDB];  // B tile, transposed (n,k)

  const int tid  = threadIdx.x;
  const int lane = tid & 31;
  const int wave = tid >> 5;
  const int l16  = lane & 15;
  const int h    = lane >> 4;       // half-wave select

  const int m0 = blockIdx.y * BM;
  const int n0 = blockIdx.x * BN;
  const int wr = wave >> 1;         // 0..3 : wave's 32-row band in BM
  const int wc = wave & 1;          // 0..1 : wave's 32-col band in BN

  v8f acc[2][2] = {};

  for (int k0 = 0; k0 < DMODEL; k0 += BK) {
    // ---- stage A tile (128x32): 16 contiguous k per thread -----------------
    {
      const int row = tid >> 1;          // 0..127
      const int kk  = (tid & 1) * 16;    // 0 or 16
      if (A_HALF) {
        const _Float16* A = (const _Float16*)Av;
        const _Float16* src = A + (size_t)(m0 + row) * DMODEL + k0 + kk;
        *(v8h*)&sA[row * LDA + kk]     = *(const v8h*)src;
        *(v8h*)&sA[row * LDA + kk + 8] = *(const v8h*)(src + 8);
        __builtin_prefetch(src + BK, 0, 3);  // speculative: OOB is dropped
      } else {
        const float* A = (const float*)Av;
        const float* src = A + (size_t)(m0 + row) * DMODEL + k0 + kk;
#pragma unroll
        for (int i = 0; i < 16; ++i) sA[row * LDA + kk + i] = (_Float16)src[i];
        __builtin_prefetch(src + BK, 0, 3);
      }
    }
    // ---- stage B tile transposed (64n x 32k): 8 n per thread ---------------
    {
      const int k  = tid >> 3;        // 0..31
      const int nn = (tid & 7) * 8;   // 0..56
      const float* src = W + (size_t)(k0 + k) * DMODEL + n0 + nn;
#pragma unroll
      for (int i = 0; i < 8; ++i) sBt[(nn + i) * LDB + k] = (_Float16)src[i];
      __builtin_prefetch(src + (size_t)BK * DMODEL, 0, 3);
    }
    __syncthreads();

    // ---- per-wave fragments + 4 WMMAs -------------------------------------
    {
      v16h af[2], bf[2];
#pragma unroll
      for (int mi = 0; mi < 2; ++mi) {
        const _Float16* ap = &sA[(wr * 32 + mi * 16 + l16) * LDA];
        af[mi] = ldfrag(ap + 8 * h, ap + 16 + 8 * h);   // K=(e%8)+8h+16*(e/8)
      }
#pragma unroll
      for (int ni = 0; ni < 2; ++ni) {
        const _Float16* bp = &sBt[(wc * 32 + ni * 16 + l16) * LDB + 8 * h];
        bf[ni] = ldfrag(bp, bp + 16);                   // K=(e%8)+8h+16*(e/8)
      }
#pragma unroll
      for (int mi = 0; mi < 2; ++mi)
#pragma unroll
        for (int ni = 0; ni < 2; ++ni)
          acc[mi][ni] = wmma_f16(af[mi], bf[ni], acc[mi][ni]);
    }
    __syncthreads();
  }

  // ---- epilogue: C layout lane N=lane%16, VGPR r -> M = r + 8h -------------
#pragma unroll
  for (int ni = 0; ni < 2; ++ni) {
    const int gn = n0 + wc * 32 + ni * 16 + l16;
    const float bv = bias[gn];
#pragma unroll
    for (int mi = 0; mi < 2; ++mi) {
      if (OUT_MODE == 2) {
        float* O = (float*)Out;
#pragma unroll
        for (int r = 0; r < 8; ++r) {
          const int gm = m0 + wr * 32 + mi * 16 + 8 * h + r;
          O[(size_t)gm * DMODEL + gn] = acc[mi][ni][r] + bv;
        }
      } else if (OUT_MODE == 0) {
        _Float16* O = (_Float16*)Out;
#pragma unroll
        for (int r = 0; r < 8; ++r) {
          const int gm = m0 + wr * 32 + mi * 16 + 8 * h + r;
          O[(size_t)gm * DMODEL + gn] = (_Float16)(acc[mi][ni][r] + bv);
        }
      } else {  // OUT_MODE == 1 : Vt[b][head][d][s]
        _Float16* O = (_Float16*)Out;
        const int hh = gn / HEADSZ;
        const int d  = gn % HEADSZ;
#pragma unroll
        for (int r = 0; r < 8; ++r) {
          const int gm = m0 + wr * 32 + mi * 16 + 8 * h + r;
          const int bb = gm / SEQ;
          const int s  = gm % SEQ;
          O[(((size_t)bb * NHEADS + hh) * HEADSZ + d) * SEQ + s] =
              (_Float16)(acc[mi][ni][r] + bv);
        }
      }
    }
  }
}

// ---------------------------------------------------------------------------
// Flash-style attention. One 16-query tile per wave; 32 keys per iteration.
// Computes S^T = K*(Q*scale)^T (softmax stats become per-lane column stats)
// and O^T = V^T * P^T, accumulating 4 head-dim tiles of 16 in fp32.
// ---------------------------------------------------------------------------
__global__ __launch_bounds__(256) void attn_kernel(const _Float16* __restrict__ Q,
                                                   const _Float16* __restrict__ K,
                                                   const _Float16* __restrict__ Vt,
                                                   _Float16* __restrict__ att) {
  const int tid  = threadIdx.x;
  const int lane = tid & 31;
  const int wave = tid >> 5;
  const int l16  = lane & 15;
  const int h    = lane >> 4;

  // block -> (b, head, 128-query chunk); wave -> 16-query tile inside chunk
  const int qcPerBH = SEQ / 128;  // 16
  const int qc   = blockIdx.x % qcPerBH;
  const int head = (blockIdx.x / qcPerBH) % NHEADS;
  const int b    = blockIdx.x / (qcPerBH * NHEADS);
  const int q    = qc * 128 + wave * 16 + l16;

  // Q fragments (B-matrix of S^T): b[e] = Q[q][32*kc + 16h + e], loaded once.
  // Fold softmax scale 1/sqrt(64) = 2^-3 into Q (exact in fp16).
  const _Float16* Qrow = Q + ((size_t)(b * SEQ + q)) * DMODEL + head * HEADSZ;
  v16h bq0 = ldfrag(Qrow + 16 * h, Qrow + 16 * h + 8) * (_Float16)0.125f;
  v16h bq1 = ldfrag(Qrow + 32 + 16 * h, Qrow + 32 + 16 * h + 8) * (_Float16)0.125f;

  const _Float16* Kbase = K + ((size_t)(b * SEQ)) * DMODEL + head * HEADSZ;
  const _Float16* Vbase = Vt + ((size_t)(b * NHEADS + head)) * HEADSZ * SEQ;

  float m = -1e30f, l = 0.0f;
  v8f co[4] = {};

  for (int kb = 0; kb < SEQ; kb += 32) {
    // ---- score tiles S^T (keys x queries), 2 tiles of 16 keys --------------
    v8f cs[2] = {};
#pragma unroll
    for (int jj = 0; jj < 2; ++jj) {
      const _Float16* Krow = Kbase + (size_t)(kb + 16 * jj + l16) * DMODEL;
      v16h ak0 = ldfrag(Krow + 16 * h, Krow + 16 * h + 8);            // d 0..31
      v16h ak1 = ldfrag(Krow + 32 + 16 * h, Krow + 40 + 16 * h);       // d 32..63
      cs[jj] = wmma_f16(ak0, bq0, cs[jj]);
      cs[jj] = wmma_f16(ak1, bq1, cs[jj]);
      __builtin_prefetch(Krow + 32 * DMODEL, 0, 3);  // speculative prefetch
    }

    // ---- online softmax over keys: column (=lane) statistics ---------------
    float p0[8], p1[8];
    float mloc = -1e30f;
#pragma unroll
    for (int r = 0; r < 8; ++r) {
      p0[r] = cs[0][r];
      p1[r] = cs[1][r];
      mloc  = fmaxf(mloc, fmaxf(p0[r], p1[r]));
    }
    mloc = fmaxf(mloc, __shfl_xor(mloc, 16, 32));  // combine half-waves
    const float newm = fmaxf(m, mloc);
    const float ef   = __expf(m - newm);
    float sum = 0.0f;
#pragma unroll
    for (int r = 0; r < 8; ++r) {
      p0[r] = __expf(p0[r] - newm);
      p1[r] = __expf(p1[r] - newm);
      sum += p0[r] + p1[r];
    }
    sum += __shfl_xor(sum, 16, 32);
    l = l * ef + sum;
    m = newm;
#pragma unroll
    for (int t = 0; t < 4; ++t) co[t] *= ef;

    // ---- build P^T B-fragment: b[e] = P^T[key=kb+e+16h][q] -----------------
    // Pack (p0[r], p1[r]) as two f16 in one dword -> one shuffle per r.
    v16h bp;
#pragma unroll
    for (int r = 0; r < 8; ++r) {
      v2h ph;
      ph[0] = (_Float16)p0[r];
      ph[1] = (_Float16)p1[r];
      const int pi = __builtin_bit_cast(int, ph);
      const v2h qo = __builtin_bit_cast(v2h, __shfl_xor(pi, 16, 32));
      bp[r]     = h ? qo[1] : ph[0];   // keys kb + r + 16h
      bp[r + 8] = h ? ph[1] : qo[0];   // keys kb + r + 8 + 16h
    }

    // ---- O^T += V^T * P^T : 4 head-dim tiles of 16 -------------------------
#pragma unroll
    for (int t = 0; t < 4; ++t) {
      const _Float16* Vrow = Vbase + (size_t)(16 * t + l16) * SEQ + kb;
      v16h av = ldfrag(Vrow + 16 * h, Vrow + 16 * h + 8);  // keys kb..kb+31
      co[t] = wmma_f16(av, bp, co[t]);
    }
  }

  // ---- normalize + store attended (merged-head row-major, fp16) ------------
  const float inv = 1.0f / l;
  _Float16* arow = att + ((size_t)(b * SEQ + q)) * DMODEL + head * HEADSZ;
#pragma unroll
  for (int t = 0; t < 4; ++t) {
    v8h ov;
#pragma unroll
    for (int r = 0; r < 8; ++r) ov[r] = (_Float16)(co[t][r] * inv);
    *(v8h*)(arow + 16 * t + 8 * h) = ov;  // d = 16t + 8h + r, contiguous
  }
}

// ---------------------------------------------------------------------------
extern "C" void kernel_launch(void* const* d_in, const int* in_sizes, int n_in,
                              void* d_out, int out_size, void* d_ws,
                              size_t ws_size, hipStream_t stream) {
  (void)in_sizes; (void)n_in; (void)out_size; (void)ws_size;
  const float* hs = (const float*)d_in[0];
  const float* WQ = (const float*)d_in[1];
  const float* bQ = (const float*)d_in[2];
  const float* WK = (const float*)d_in[3];
  const float* bK = (const float*)d_in[4];
  const float* WV = (const float*)d_in[5];
  const float* bV = (const float*)d_in[6];
  const float* WY = (const float*)d_in[7];
  const float* bY = (const float*)d_in[8];

  char* ws = (char*)d_ws;
  const size_t matBytes = (size_t)MROWS * DMODEL * sizeof(_Float16);  // 6.3 MB
  _Float16* Qh  = (_Float16*)(ws + 0 * matBytes);
  _Float16* Kh  = (_Float16*)(ws + 1 * matBytes);
  _Float16* Vth = (_Float16*)(ws + 2 * matBytes);  // [b][h][d][s]
  _Float16* Ah  = (_Float16*)(ws + 3 * matBytes);  // attended, merged heads

  dim3 gg(DMODEL / 64, MROWS / 128);  // (12, 32)
  gemm16<false, 0><<<gg, 256, 0, stream>>>((const void*)hs, WQ, bQ, (void*)Qh);
  gemm16<false, 0><<<gg, 256, 0, stream>>>((const void*)hs, WK, bK, (void*)Kh);
  gemm16<false, 1><<<gg, 256, 0, stream>>>((const void*)hs, WV, bV, (void*)Vth);

  const int nBlocks = BATCH * NHEADS * (SEQ / 128);  // 384
  attn_kernel<<<nBlocks, 256, 0, stream>>>(Qh, Kh, Vth, Ah);

  gemm16<true, 2><<<gg, 256, 0, stream>>>((const void*)Ah, WY, bY, d_out);
}
